// Att_LSTM_67989332296335
// MI455X (gfx1250) — matrix operand order repaired and checked by reading, not compiled, not run on hardware
//
#include <hip/hip_runtime.h>
#include <hip/hip_bf16.h>
#include <math.h>

// Problem dims (reference): S=1024, B=32, E=256, H=512, AH=256, V=32000, T=20
#define S_LEN  1024
#define BATCH  32
#define EDIM   256
#define HDIM   512
#define AHDIM  256
#define TDIM   20
#define KCAT   768           // E + H (LSTM concatenated K)
#define NGATE  2048          // 4*H
#define ASTRIDE 784          // LDS A-buffer row stride (bank skew, keeps 16B align)

typedef __attribute__((ext_vector_type(16))) __bf16 v16bf;
typedef __attribute__((ext_vector_type(8)))  __bf16 v8bf;
typedef __attribute__((ext_vector_type(8)))  float  v8f;

union V16U { v16bf v; v8bf h[2]; };

__device__ __forceinline__ v8f wmma_bf16(v16bf a, v16bf b, v8f c) {
  // D = A(16x32) * B(32x16) + C(16x16 f32)  -> v_wmma_f32_16x16x32_bf16
  return __builtin_amdgcn_wmma_f32_16x16x32_bf16(false, a, false, b, (short)0, c,
                                                 false, false);
}

// A fragment (16x32, row-major source, row stride in elements).
// Lane l<16 holds row M=l, K = kb+0..7 (lo) and kb+16..23 (hi);
// lane l>=16 holds row M=l-16, K = kb+8..15 and kb+24..31. (ISA 7.12.2)
__device__ __forceinline__ v16bf load_a_frag(const __bf16* base, int stride,
                                             int rowbase, int kbase, int lane) {
  int r = rowbase + (lane & 15);
  int k = kbase + ((lane >> 4) << 3);
  const __bf16* p = base + (size_t)r * stride + k;
  V16U u;
  u.h[0] = *(const v8bf*)(p);
  u.h[1] = *(const v8bf*)(p + 16);
  return u.v;
}

// B fragment (32x16). Source is N-major (weight row n = column n of B),
// so lane l holds column N = l&15, K = kb+8*(l>=16)+0..7 and kb+16+8*(l>=16)+0..7.
__device__ __forceinline__ v16bf load_b_frag(const __bf16* base, int stride,
                                             int nbase, int kbase, int lane) {
  int n = nbase + (lane & 15);
  int k = kbase + ((lane >> 4) << 3);
  const __bf16* p = base + (size_t)n * stride + k;
  V16U u;
  u.h[0] = *(const v8bf*)(p);
  u.h[1] = *(const v8bf*)(p + 16);
  return u.v;
}

__device__ __forceinline__ float sigm_f(float x) {
  return 1.f / (1.f + __expf(-x));
}
__device__ __forceinline__ float tanh_f(float x) {
  x = fminf(fmaxf(x, -15.f), 15.f);
  float t = __expf(2.f * x);
  return (t - 1.f) / (t + 1.f);
}

// ---------------- prep kernels ----------------

__global__ __launch_bounds__(256)
void embed_gather_k(const int* __restrict__ x, const float* __restrict__ embed,
                    __bf16* __restrict__ emb) {
  int i  = blockIdx.x * 256 + threadIdx.x;   // over S*B*E
  int e  = i & (EDIM - 1);
  int sb = i >> 8;
  int tok = x[sb];
  emb[i] = (__bf16)embed[(size_t)tok * EDIM + e];
}

__global__ __launch_bounds__(256)
void prep_wcat_k(const float* __restrict__ W_ih, const float* __restrict__ W_hh,
                 __bf16* __restrict__ Wcat) {
  int i = blockIdx.x * 256 + threadIdx.x;    // over 2048*768
  int row = i / KCAT;
  int k   = i - row * KCAT;
  float v = (k < EDIM) ? W_ih[(size_t)row * EDIM + k]
                       : W_hh[(size_t)row * HDIM + (k - EDIM)];
  Wcat[i] = (__bf16)v;
}

__global__ __launch_bounds__(256)
void prep_bsum_k(const float* __restrict__ b_ih, const float* __restrict__ b_hh,
                 float* __restrict__ bsum) {
  int i = blockIdx.x * 256 + threadIdx.x;    // over 2048
  bsum[i] = b_ih[i] + b_hh[i];
}

__global__ __launch_bounds__(256)
void prep_wg_k(const float* __restrict__ W_g, __bf16* __restrict__ Wg) {
  int i = blockIdx.x * 256 + threadIdx.x;    // over 256*1024
  Wg[i] = (__bf16)W_g[i];
}

__global__ __launch_bounds__(256)
void prep_wtag_k(const float* __restrict__ W_tag, __bf16* __restrict__ Wt) {
  int i = blockIdx.x * 256 + threadIdx.x;    // over 32*256 (rows 20..31 zero pad)
  int row = i >> 8;
  int col = i & 255;
  Wt[i] = (row < TDIM) ? (__bf16)W_tag[(size_t)row * AHDIM + col] : (__bf16)0.f;
}

// ---------------- LSTM (sequential over S; one WGP; c & Gsum in VGPRs) ----------------

#define LSTM_THREADS 512   // 16 waves; wave w owns column tiles {w, w+16} of H

__global__ __launch_bounds__(LSTM_THREADS)
void lstm_k(const __bf16* __restrict__ emb,   // (S,B,E) bf16
            const __bf16* __restrict__ Wcat,  // (4H, E+H) bf16, n-major
            const float*  __restrict__ bsum,  // (4H) = b_ih + b_hh
            __bf16* __restrict__ hs,          // out (S,B,H) bf16
            __bf16* __restrict__ Gsum)        // out (B,H) bf16 = sum_s hs[s]
{
  __shared__ __bf16 Abuf[BATCH * ASTRIDE];    // [x_t (256) | h_{t-1} (512)] rows=batch
  const int tid  = threadIdx.x;
  const int lane = tid & 31;
  const int wv   = tid >> 5;                  // 0..15
  const int colq = lane & 15;

  for (int i = tid; i < BATCH * ASTRIDE; i += LSTM_THREADS) Abuf[i] = (__bf16)0.f;

  float bias[2][4];
  #pragma unroll
  for (int cg = 0; cg < 2; ++cg) {
    int col = (wv + cg * 16) * 16 + colq;
    #pragma unroll
    for (int g = 0; g < 4; ++g) bias[cg][g] = bsum[g * HDIM + col];
  }

  v8f cst[2][2] = {};   // cell state  [col-group][m-tile], lives in VGPRs
  v8f gsm[2][2] = {};   // sum_s h     (== G, since softmax over singleton axis == 1)
  v8f hreg[2][2];
  __syncthreads();

  for (int s = 0; s < S_LEN; ++s) {
    // stage x_t into A-buffer cols [0,256): 512 threads x 16 bf16
    {
      int off = tid * 16;
      int b = off >> 8, e = off & 255;
      const __bf16* src = emb + ((size_t)(s * BATCH + b)) * EDIM + e;
      *(v8bf*)&Abuf[b * ASTRIDE + e]     = *(const v8bf*)(src);
      *(v8bf*)&Abuf[b * ASTRIDE + e + 8] = *(const v8bf*)(src + 8);
    }
    __syncthreads();   // staging + previous step's h writes visible

    #pragma unroll 1
    for (int cg = 0; cg < 2; ++cg) {
      const int ct  = wv + cg * 16;         // column tile of H, 0..31
      const int col = ct * 16 + colq;       // output column 0..511
      v8f acc[4][2];
      #pragma unroll
      for (int g = 0; g < 4; ++g)
        #pragma unroll
        for (int m = 0; m < 2; ++m)
          #pragma unroll
          for (int q = 0; q < 8; ++q) acc[g][m][q] = bias[cg][g];

      #pragma unroll 2
      for (int kt = 0; kt < KCAT / 32; ++kt) {          // 24 k-steps
        int kb = kt * 32;
        v16bf a0 = load_a_frag(Abuf, ASTRIDE,  0, kb, lane);
        v16bf a1 = load_a_frag(Abuf, ASTRIDE, 16, kb, lane);
        #pragma unroll
        for (int g = 0; g < 4; ++g) {
          v16bf bm = load_b_frag(Wcat, KCAT, g * HDIM + ct * 16, kb, lane);
          acc[g][0] = wmma_bf16(a0, bm, acc[g][0]);
          acc[g][1] = wmma_bf16(a1, bm, acc[g][1]);
        }
      }

      // gate fusion (torch order i,f,g,o); c and Gsum stay in registers
      #pragma unroll
      for (int m = 0; m < 2; ++m) {
        #pragma unroll
        for (int q = 0; q < 8; ++q) {
          float ig = sigm_f(acc[0][m][q]);
          float fg = sigm_f(acc[1][m][q]);
          float gg = tanh_f(acc[2][m][q]);
          float og = sigm_f(acc[3][m][q]);
          float cv = fg * cst[cg][m][q] + ig * gg;
          cst[cg][m][q] = cv;
          float hv = og * tanh_f(cv);
          gsm[cg][m][q] += hv;
          hreg[cg][m][q] = hv;
          int row = m * 16 + q + ((lane >> 4) << 3);    // batch index
          hs[((size_t)(s * BATCH + row)) * HDIM + col] = (__bf16)hv;
        }
      }
    }

    __syncthreads();   // all waves done reading A before h_t overwrites it
    #pragma unroll
    for (int cg = 0; cg < 2; ++cg) {
      int col = (wv + cg * 16) * 16 + colq;
      #pragma unroll
      for (int m = 0; m < 2; ++m)
        #pragma unroll
        for (int q = 0; q < 8; ++q) {
          int row = m * 16 + q + ((lane >> 4) << 3);
          Abuf[row * ASTRIDE + EDIM + col] = (__bf16)hreg[cg][m][q];
        }
    }
  }

  // G = sum_s hs (attention weights are identically 1.0)
  #pragma unroll
  for (int cg = 0; cg < 2; ++cg) {
    int col = (wv + cg * 16) * 16 + colq;
    #pragma unroll
    for (int m = 0; m < 2; ++m)
      #pragma unroll
      for (int q = 0; q < 8; ++q) {
        int row = m * 16 + q + ((lane >> 4) << 3);
        Gsum[(size_t)row * HDIM + col] = (__bf16)gsm[cg][m][q];
      }
  }
}

// ---------------- Z = tanh([G | hs] @ W_g^T + b_g) ----------------

__global__ __launch_bounds__(256)
void zgemm_k(const __bf16* __restrict__ G,    // (B,H) bf16
             const __bf16* __restrict__ hs,   // (S,B,H) bf16
             const __bf16* __restrict__ Wg,   // (AH, 2H) bf16, n-major
             const float*  __restrict__ b_g,  // (AH)
             __bf16* __restrict__ Z)          // (S*B, AH) bf16
{
  int lane = threadIdx.x & 31;
  int wv   = threadIdx.x >> 5;
  int mt   = blockIdx.x * 8 + wv;             // 0..2047 (16-row tiles of S*B)
  int nt   = blockIdx.y;                      // 0..15
  int s      = mt >> 1;
  int bbase  = (mt & 1) << 4;

  v8f acc = {};
  #pragma unroll 4
  for (int kt = 0; kt < HDIM / 32; ++kt) {    // G half (k in [0,512))
    v16bf a = load_a_frag(G, HDIM, bbase, kt * 32, lane);
    v16bf b = load_b_frag(Wg, 2 * HDIM, nt * 16, kt * 32, lane);
    acc = wmma_bf16(a, b, acc);
  }
  const __bf16* hsrow = hs + (size_t)s * BATCH * HDIM;
  #pragma unroll 4
  for (int kt = 0; kt < HDIM / 32; ++kt) {    // hs half (k in [512,1024))
    v16bf a = load_a_frag(hsrow, HDIM, bbase, kt * 32, lane);
    v16bf b = load_b_frag(Wg, 2 * HDIM, nt * 16, HDIM + kt * 32, lane);
    acc = wmma_bf16(a, b, acc);
  }

  int nc = nt * 16 + (lane & 15);
  float bias = b_g[nc];
  #pragma unroll
  for (int q = 0; q < 8; ++q) {
    int row = bbase + q + ((lane >> 4) << 3);
    Z[((size_t)(s * BATCH + row)) * AHDIM + nc] = (__bf16)tanh_f(acc[q] + bias);
  }
}

// ---------------- tag = Z @ W_tag^T + b_tag  (N padded 20 -> 32) ----------------

__global__ __launch_bounds__(256)
void tag_k(const __bf16* __restrict__ Z,      // (S*B, AH)
           const __bf16* __restrict__ Wt,     // (32, AH), rows 20..31 zero
           const float*  __restrict__ b_tag,  // (T)
           float* __restrict__ tagbuf)        // (S, T, B) f32
{
  int lane = threadIdx.x & 31;
  int wv   = threadIdx.x >> 5;
  int mt   = blockIdx.x * 8 + wv;             // 0..2047
  int s      = mt >> 1;
  int bbase  = (mt & 1) << 4;
  const __bf16* Zrow = Z + (size_t)s * BATCH * AHDIM;

  v8f acc0 = {}, acc1 = {};
  #pragma unroll
  for (int kt = 0; kt < AHDIM / 32; ++kt) {   // 8 k-steps
    v16bf a  = load_a_frag(Zrow, AHDIM, bbase, kt * 32, lane);
    v16bf b0 = load_b_frag(Wt, AHDIM,  0, kt * 32, lane);
    v16bf b1 = load_b_frag(Wt, AHDIM, 16, kt * 32, lane);
    acc0 = wmma_bf16(a, b0, acc0);
    acc1 = wmma_bf16(a, b1, acc1);
  }

  int tq = lane & 15;                         // tag index 0..15 (acc0) / 16..31 (acc1)
  #pragma unroll
  for (int q = 0; q < 8; ++q) {
    int row = bbase + q + ((lane >> 4) << 3);
    tagbuf[((size_t)s * TDIM + tq) * BATCH + row] = acc0[q] + b_tag[tq];
    int t1 = 16 + tq;
    if (t1 < TDIM)
      tagbuf[((size_t)s * TDIM + t1) * BATCH + row] = acc1[q] + b_tag[t1];
  }
}

// ---------------- log_softmax over the BATCH axis (reference dim=1) ----------------

__global__ __launch_bounds__(256)
void lsm_k(const float* __restrict__ tagbuf,  // (S, T, B)
           float* __restrict__ out)           // (S, B, T)
{
  int gw   = (blockIdx.x * 256 + threadIdx.x) >> 5;   // (s,t) pair, one per wave
  int lane = threadIdx.x & 31;                         // lane = batch index
  if (gw >= S_LEN * TDIM) return;
  float v = tagbuf[(size_t)gw * BATCH + lane];
  float m = v;
  #pragma unroll
  for (int o = 16; o > 0; o >>= 1) m = fmaxf(m, __shfl_xor(m, o, 32));
  float e = __expf(v - m);
  float sum = e;
  #pragma unroll
  for (int o = 16; o > 0; o >>= 1) sum += __shfl_xor(sum, o, 32);
  int s = gw / TDIM;
  int t = gw - s * TDIM;
  out[((size_t)s * BATCH + lane) * TDIM + t] = (v - m) - __logf(sum);
}

// ---------------- launch ----------------

extern "C" void kernel_launch(void* const* d_in, const int* in_sizes, int n_in,
                              void* d_out, int out_size, void* d_ws, size_t ws_size,
                              hipStream_t stream) {
  (void)in_sizes; (void)n_in; (void)out_size; (void)ws_size;
  const int*   x     = (const int*)  d_in[0];
  const float* embed = (const float*)d_in[1];
  const float* W_ih  = (const float*)d_in[2];
  const float* W_hh  = (const float*)d_in[3];
  const float* b_ih  = (const float*)d_in[4];
  const float* b_hh  = (const float*)d_in[5];
  // d_in[6]=w_sim, d_in[7]=b_sim are dead: softmax over a singleton axis == 1.0
  const float* W_g   = (const float*)d_in[8];
  const float* b_g   = (const float*)d_in[9];
  const float* W_tag = (const float*)d_in[10];
  const float* b_tag = (const float*)d_in[11];
  float* out = (float*)d_out;

  char* p = (char*)d_ws;
  auto take = [&](size_t bytes) -> void* {
    void* r = (void*)p;
    p += (bytes + 255) & ~(size_t)255;
    return r;
  };
  __bf16* emb_bf  = (__bf16*)take((size_t)S_LEN * BATCH * EDIM * 2);   // 16.8 MB
  __bf16* Wcat_bf = (__bf16*)take((size_t)NGATE * KCAT * 2);           //  3.1 MB
  float*  bsum    = (float*) take((size_t)NGATE * 4);
  __bf16* hs_bf   = (__bf16*)take((size_t)S_LEN * BATCH * HDIM * 2);   // 33.6 MB
  __bf16* G_bf    = (__bf16*)take((size_t)BATCH * HDIM * 2);
  __bf16* Wg_bf   = (__bf16*)take((size_t)AHDIM * 2 * HDIM * 2);
  __bf16* Z_bf    = (__bf16*)take((size_t)S_LEN * BATCH * AHDIM * 2);  // 16.8 MB
  __bf16* Wt_bf   = (__bf16*)take((size_t)32 * AHDIM * 2);
  float*  tagbuf  = (float*) take((size_t)S_LEN * TDIM * BATCH * 4);   //  2.6 MB

  embed_gather_k<<<(S_LEN * BATCH * EDIM) / 256, 256, 0, stream>>>(x, embed, emb_bf);
  prep_wcat_k<<<(NGATE * KCAT) / 256, 256, 0, stream>>>(W_ih, W_hh, Wcat_bf);
  prep_bsum_k<<<NGATE / 256, 256, 0, stream>>>(b_ih, b_hh, bsum);
  prep_wg_k<<<(AHDIM * 2 * HDIM) / 256, 256, 0, stream>>>(W_g, Wg_bf);
  prep_wtag_k<<<(32 * AHDIM) / 256, 256, 0, stream>>>(W_tag, Wt_bf);

  lstm_k<<<1, LSTM_THREADS, 0, stream>>>(emb_bf, Wcat_bf, bsum, hs_bf, G_bf);

  zgemm_k<<<dim3(256, 16), 256, 0, stream>>>(G_bf, hs_bf, Wg_bf, b_g, Z_bf);
  tag_k<<<256, 256, 0, stream>>>(Z_bf, Wt_bf, b_tag, tagbuf);
  lsm_k<<<(S_LEN * TDIM) / 8, 256, 0, stream>>>(tagbuf, out);
}
